// MolGATMesPassing_64046552318003
// MI455X (gfx1250) — matrix operand order repaired and checked by reading, not compiled, not run on hardware
//
#include <hip/hip_runtime.h>

// ---------------------------------------------------------------------------
// MolGAT message passing, fused for MI455X (gfx1250, wave32, WMMA).
//
// context  = elu(softmax(nem * a^T + mask) @ leaky(upv@W_lin+b))   [8192,128]
// new_edge = leaky(upv@W_e+b)                                      [16384,128]
// a        = leaky(concat(edge,pos)@W_att+b)                       [16384]
//
// Single pass over the 1 GB nem/mask stream (HBM roofline ~44us) with online
// softmax fused into bf16 WMMA (v_wmma_f32_16x16x32_bf16), K-split x4 for
// memory-level parallelism (2048 waves), partials merged flash-style.
// ---------------------------------------------------------------------------

typedef __bf16 bf16_t;
typedef __attribute__((ext_vector_type(16))) __bf16 v16bf;
typedef __attribute__((ext_vector_type(8)))  float  v8f;

union V16 { v16bf bf; uint4 u[2]; };

#define N_NODES 8192
#define N_EDGES 16384
#define C_DIM   128
#define E_DIM   128
#define IN_DIM  272      // 2*128 + 16
#define K_PAD   288      // IN_DIM padded to 9 * 32
#define KB_EDGE 9        // 288 / 32
#define NCHUNK  4        // K-split factor for the big GEMM
#define CHUNK   (N_EDGES / NCHUNK)   // 4096 edges per chunk

__device__ __forceinline__ float leaky(float x) { return x > 0.f ? x : 0.01f * x; }
__device__ __forceinline__ float elu_f(float x) { return x > 0.f ? x : __expf(x) - 1.f; }

// ---------------------------------------------------------------------------
// K0: pack [W_e | W_lin] (272x128 each, f32) into bf16 WMMA-B tile layout.
// Tile (kb, nt): 32 lanes x 16 bf16.  Lanes 0-15 hold K = kb*32 + 0..7,16..23
// (2 per VGPR), lanes 16-31 hold K = kb*32 + 8..15,24..31; N = nt*16 + lane%16.
// ---------------------------------------------------------------------------
__global__ void k0_pack(const float* __restrict__ W_lin,
                        const float* __restrict__ W_e,
                        bf16_t* __restrict__ Wpk) {
  int tid = blockIdx.x * blockDim.x + threadIdx.x;   // 0 .. 9*16*32-1
  if (tid >= KB_EDGE * 16 * 32) return;
  int kb   = tid / 512;
  int rem  = tid % 512;
  int nt   = rem / 32;
  int lane = rem % 32;
  int h    = lane >> 4;
  int n    = nt * 16 + (lane & 15);

  bf16_t vals[16];
#pragma unroll
  for (int i = 0; i < 16; ++i) {
    int K = kb * 32 + h * 8 + (i < 8 ? i : i + 8);
    float x = 0.f;
    if (K < IN_DIM)
      x = (n < E_DIM) ? W_e[K * E_DIM + n] : W_lin[K * C_DIM + (n - E_DIM)];
    vals[i] = (bf16_t)x;
  }
  bf16_t* dst = Wpk + (size_t)tid * 16;
  uint4 u0, u1;
  __builtin_memcpy(&u0, vals, 16);
  __builtin_memcpy(&u1, vals + 8, 16);
  *(uint4*)dst        = u0;
  *(uint4*)(dst + 8)  = u1;
}

// ---------------------------------------------------------------------------
// K1: per-wave 16-edge tile.  Stage bf16 upv panel (16 x 288, zero padded) in
// LDS, compute both GEMMs with v_wmma_f32_16x16x32_bf16.  All B tiles for one
// N-tile are hoisted into a register array so the 18 b128 loads issue as one
// clause batch ahead of 9 back-to-back WMMAs (only ~1 wave/SIMD here).
// Block = 128 threads = 4 waves = 64 edges.  Grid = 256.
// ---------------------------------------------------------------------------
__global__ void __launch_bounds__(128)
k1_edge(const float* __restrict__ u_feat, const float* __restrict__ v_feat,
        const float* __restrict__ e_feat, const float* __restrict__ p_feat,
        const float* __restrict__ b_lin,  const float* __restrict__ b_e,
        const float* __restrict__ W_att,  const float* __restrict__ b_att,
        const bf16_t* __restrict__ Wpk,
        float* __restrict__ out_edge,     // [16384,128]
        bf16_t* __restrict__ NFT,         // [128,16384] transposed bf16
        float* __restrict__ a_out) {      // [16384]
  __shared__ alignas(16) bf16_t s_upv[4][16][K_PAD];

  const int lane = threadIdx.x & 31;
  const int w    = threadIdx.x >> 5;
  const int r    = lane & 15;             // row / column-within-tile
  const int h    = lane >> 4;             // half-wave
  const int e0   = blockIdx.x * 64 + w * 16;

  // ---- stage upv = [u | pos | v] as bf16 into LDS (pad K 272..287 = 0) ----
  for (int idx = lane; idx < 16 * K_PAD; idx += 32) {
    int row = idx / K_PAD, k = idx % K_PAD;
    float val;
    if (k < 128)       val = u_feat[(size_t)(e0 + row) * 128 + k];
    else if (k < 144)  val = p_feat[(size_t)(e0 + row) * 16 + (k - 128)];
    else if (k < 272)  val = v_feat[(size_t)(e0 + row) * 128 + (k - 144)];
    else               val = 0.f;
    s_upv[w][row][k] = (bf16_t)val;
  }

  // ---- attention scalar: a[e] = leaky(concat(edge,pos) . W_att + b) ----
  {
    float part = 0.f;
    for (int j = h * 72; j < h * 72 + 72; ++j) {
      float x = (j < 128) ? e_feat[(size_t)(e0 + r) * 128 + j]
                          : p_feat[(size_t)(e0 + r) * 16 + (j - 128)];
      part += x * W_att[j];
    }
    part += __shfl_xor(part, 16, 32);
    if (h == 0) a_out[e0 + r] = leaky(part + b_att[0]);
  }

  __syncthreads();

  // ---- A-operand panels: 2 x ds_load_b128 per 16x32 K-block ----
  V16 va[KB_EDGE];
#pragma unroll
  for (int kb = 0; kb < KB_EDGE; ++kb) {
    const bf16_t* ap = &s_upv[w][r][kb * 32 + h * 8];
    va[kb].u[0] = *(const uint4*)ap;          // K 0..7  (lo) / 8..15  (hi)
    va[kb].u[1] = *(const uint4*)(ap + 16);   // K 16..23 (lo)/ 24..31 (hi)
  }

  // ---- 16 N-tiles: nt<8 -> W_e (new_edge), nt>=8 -> W_lin (neighbor) ----
  for (int nt = 0; nt < 16; ++nt) {
    int n = nt * 16 + r;
    float bias = (n < E_DIM) ? b_e[n] : b_lin[n - E_DIM];
    v8f c;
#pragma unroll
    for (int j = 0; j < 8; ++j) c[j] = bias;

    V16 vb[KB_EDGE];                    // hoist: 18 loads, then 9 wmma
#pragma unroll
    for (int kb = 0; kb < KB_EDGE; ++kb) {
      const bf16_t* bp = Wpk + ((size_t)(kb * 16 + nt) * 32 + lane) * 16;
      vb[kb].u[0] = *(const uint4*)bp;
      vb[kb].u[1] = *(const uint4*)(bp + 8);
    }
#pragma unroll
    for (int kb = 0; kb < KB_EDGE; ++kb)
      c = __builtin_amdgcn_wmma_f32_16x16x32_bf16(
              false, va[kb].bf, false, vb[kb].bf, (short)0, c, false, false);

    if (nt < 8) {               // new_edge_features, f32
#pragma unroll
      for (int j = 0; j < 8; ++j)
        out_edge[(size_t)(e0 + j + 8 * h) * E_DIM + n] = leaky(c[j]);
    } else {                    // neighbor features -> transposed bf16
#pragma unroll
      for (int j = 0; j < 8; ++j)
        NFT[(size_t)(n - E_DIM) * N_EDGES + (e0 + j + 8 * h)] =
            (bf16_t)leaky(c[j]);
    }
  }
}

// ---------------------------------------------------------------------------
// K2: fused online-softmax + GEMM over one K chunk (flash-decoding split).
// One wave = 16 node rows x 4096 edges; 128 K-blocks of 32.  nem/mask loaded
// straight into the WMMA A lane layout, exp-weights packed to bf16, 8 wmma
// per block against NFT.  Writes unnormalized partials (acc, m, s).
// Block = 128 threads (4 waves = the 4 chunks of one row tile), grid = 512.
// ---------------------------------------------------------------------------
__global__ void __launch_bounds__(128)
k2_partial(const float* __restrict__ nem, const float* __restrict__ mask,
           const float* __restrict__ a_in, const bf16_t* __restrict__ NFT,
           float* __restrict__ pacc,   // [512][4][16][128]
           float* __restrict__ pm,     // [512][4][16]
           float* __restrict__ ps) {   // [512][4][16]
  const int lane = threadIdx.x & 31;
  const int c    = threadIdx.x >> 5;       // chunk 0..3
  const int r    = lane & 15;
  const int h    = lane >> 4;
  const int t    = blockIdx.x;             // row tile 0..511
  const int m0   = t * 16;
  const int ebeg = c * CHUNK;

  v8f acc[8];
#pragma unroll
  for (int nt = 0; nt < 8; ++nt)
#pragma unroll
    for (int j = 0; j < 8; ++j) acc[nt][j] = 0.f;

  float m_run = -1e30f;
  float s_run = 0.f;

  const float* nrow = nem  + (size_t)(m0 + r) * N_EDGES;
  const float* mrow = mask + (size_t)(m0 + r) * N_EDGES;

#pragma unroll 2
  for (int kb = 0; kb < CHUNK / 32; ++kb) {
    const int e0    = ebeg + kb * 32;
    const int kbase = e0 + h * 8;     // lane's K set: kbase+0..7, +16..23

    if (kb + 8 < CHUNK / 32) {        // stream 8 blocks (1 KB/row) ahead
      __builtin_prefetch(nrow + e0 + 256, 0, 1);
      __builtin_prefetch(mrow + e0 + 256, 0, 1);
    }

    // w = nem * a + mask, directly in A layout (16 values / lane)
    float4 n0 = *(const float4*)(nrow + kbase);
    float4 n1 = *(const float4*)(nrow + kbase + 4);
    float4 n2 = *(const float4*)(nrow + kbase + 16);
    float4 n3 = *(const float4*)(nrow + kbase + 20);
    float4 k0 = *(const float4*)(mrow + kbase);
    float4 k1 = *(const float4*)(mrow + kbase + 4);
    float4 k2 = *(const float4*)(mrow + kbase + 16);
    float4 k3 = *(const float4*)(mrow + kbase + 20);
    float4 a0 = *(const float4*)(a_in + kbase);
    float4 a1 = *(const float4*)(a_in + kbase + 4);
    float4 a2 = *(const float4*)(a_in + kbase + 16);
    float4 a3 = *(const float4*)(a_in + kbase + 20);

    float p[16];
    p[0]=fmaf(n0.x,a0.x,k0.x);  p[1]=fmaf(n0.y,a0.y,k0.y);
    p[2]=fmaf(n0.z,a0.z,k0.z);  p[3]=fmaf(n0.w,a0.w,k0.w);
    p[4]=fmaf(n1.x,a1.x,k1.x);  p[5]=fmaf(n1.y,a1.y,k1.y);
    p[6]=fmaf(n1.z,a1.z,k1.z);  p[7]=fmaf(n1.w,a1.w,k1.w);
    p[8]=fmaf(n2.x,a2.x,k2.x);  p[9]=fmaf(n2.y,a2.y,k2.y);
    p[10]=fmaf(n2.z,a2.z,k2.z); p[11]=fmaf(n2.w,a2.w,k2.w);
    p[12]=fmaf(n3.x,a3.x,k3.x); p[13]=fmaf(n3.y,a3.y,k3.y);
    p[14]=fmaf(n3.z,a3.z,k3.z); p[15]=fmaf(n3.w,a3.w,k3.w);

    // online softmax: row max (pair across half-waves), rescale, exp, sum
    float bm = p[0];
#pragma unroll
    for (int i = 1; i < 16; ++i) bm = fmaxf(bm, p[i]);
    bm = fmaxf(bm, __shfl_xor(bm, 16, 32));

    float m_new = fmaxf(m_run, bm);
    float fs    = __expf(m_run - m_new);
    m_run = m_new;

    float bs = 0.f;
#pragma unroll
    for (int i = 0; i < 16; ++i) { p[i] = __expf(p[i] - m_new); bs += p[i]; }
    bs += __shfl_xor(bs, 16, 32);
    s_run = s_run * fs + bs;

    // rescale accumulators: row factor broadcast via ds_bpermute
#pragma unroll
    for (int j = 0; j < 8; ++j) {
      float fj = __shfl(fs, j + (h ? 8 : 0), 32);
#pragma unroll
      for (int nt = 0; nt < 8; ++nt) acc[nt][j] *= fj;
    }

    // pack exp-weights to bf16 A operand
    V16 pa;
#pragma unroll
    for (int i = 0; i < 16; ++i) pa.bf[i] = (bf16_t)p[i];

    // hoist all 8 B tiles (16 x b128), then 8 back-to-back wmma
    V16 vb[8];
#pragma unroll
    for (int nt = 0; nt < 8; ++nt) {
      const bf16_t* bp = NFT + (size_t)(nt * 16 + r) * N_EDGES + e0 + h * 8;
      vb[nt].u[0] = *(const uint4*)bp;
      vb[nt].u[1] = *(const uint4*)(bp + 16);
    }
#pragma unroll
    for (int nt = 0; nt < 8; ++nt)
      acc[nt] = __builtin_amdgcn_wmma_f32_16x16x32_bf16(
                    false, pa.bf, false, vb[nt].bf, (short)0, acc[nt], false, false);
  }

  // ---- store unnormalized partials ----
  const int base = (t * NCHUNK + c) * 16;
#pragma unroll
  for (int j = 0; j < 8; ++j) {
    int rl = j + 8 * h;
#pragma unroll
    for (int nt = 0; nt < 8; ++nt)
      pacc[(size_t)(base + rl) * C_DIM + nt * 16 + r] = acc[nt][j];
  }
  if (h == 0) { pm[base + r] = m_run; ps[base + r] = s_run; }
}

// ---------------------------------------------------------------------------
// K3: merge the 4 K-chunk partials per row (flash combine), ELU, store.
// ---------------------------------------------------------------------------
__global__ void __launch_bounds__(256)
k3_combine(const float* __restrict__ pacc, const float* __restrict__ pm,
           const float* __restrict__ ps, float* __restrict__ out_ctx) {
  int tid = blockIdx.x * 256 + threadIdx.x;     // 0 .. 8192*128-1
  int row = tid >> 7, col = tid & 127;
  int t = row >> 4, rl = row & 15;

  float mv[NCHUNK];
  float M = -1e30f;
#pragma unroll
  for (int c = 0; c < NCHUNK; ++c) {
    mv[c] = pm[(t * NCHUNK + c) * 16 + rl];
    M = fmaxf(M, mv[c]);
  }
  float S = 0.f, V = 0.f;
#pragma unroll
  for (int c = 0; c < NCHUNK; ++c) {
    float ef = __expf(mv[c] - M);
    S += ps[(t * NCHUNK + c) * 16 + rl] * ef;
    V += pacc[(size_t)((t * NCHUNK + c) * 16 + rl) * C_DIM + col] * ef;
  }
  out_ctx[(size_t)row * C_DIM + col] = elu_f(V / S);
}

// ---------------------------------------------------------------------------
extern "C" void kernel_launch(void* const* d_in, const int* in_sizes, int n_in,
                              void* d_out, int out_size, void* d_ws,
                              size_t ws_size, hipStream_t stream) {
  (void)in_sizes; (void)n_in; (void)out_size; (void)ws_size;
  const float* u     = (const float*)d_in[0];
  const float* v     = (const float*)d_in[1];
  const float* edg   = (const float*)d_in[2];
  const float* pos   = (const float*)d_in[3];
  const float* nem   = (const float*)d_in[4];
  const float* mask  = (const float*)d_in[5];
  const float* W_lin = (const float*)d_in[6];
  const float* b_lin = (const float*)d_in[7];
  const float* W_e   = (const float*)d_in[8];
  const float* b_e   = (const float*)d_in[9];
  const float* W_att = (const float*)d_in[10];
  const float* b_att = (const float*)d_in[11];

  float* out_ctx  = (float*)d_out;                       // [8192,128]
  float* out_edge = out_ctx + (size_t)N_NODES * C_DIM;   // [16384,128]

  char* ws = (char*)d_ws;
  bf16_t* Wpk  = (bf16_t*)ws;                      //   147456 B packed weights
  float*  aw   = (float*)(ws + 147456);            //    65536 B attn scalars
  bf16_t* NFT  = (bf16_t*)(ws + 212992);           //  4 MB neighbor^T bf16
  float*  pacc = (float*)(ws + 4407296);           // 16 MB chunk partial acc
  float*  pm   = (float*)(ws + 21184512);          //   128 KB partial max
  float*  psum = (float*)(ws + 21315584);          //   128 KB partial sum

  hipLaunchKernelGGL(k0_pack, dim3(18), dim3(256), 0, stream, W_lin, W_e, Wpk);
  hipLaunchKernelGGL(k1_edge, dim3(256), dim3(128), 0, stream,
                     u, v, edg, pos, b_lin, b_e, W_att, b_att,
                     Wpk, out_edge, NFT, aw);
  hipLaunchKernelGGL(k2_partial, dim3(512), dim3(128), 0, stream,
                     nem, mask, aw, NFT, pacc, pm, psum);
  hipLaunchKernelGGL(k3_combine, dim3(4096), dim3(256), 0, stream,
                     pacc, pm, psum, out_ctx);
}
